// AdditiveAttention_50491635532087
// MI455X (gfx1250) — matrix operand order repaired and checked by reading, not compiled, not run on hardware
//
#include <hip/hip_runtime.h>
#include <hip/hip_bf16.h>
#include <math.h>

#define B_   32
#define T_   4096
#define H_   512
#define NEGV -1000000000.0f

#define TTILE 64               // T-rows per block in scores kernel
#define LSTRF (H_ + 4)         // LDS row stride in f32 elems (16B pad -> 4-bank shift/row)

typedef __attribute__((ext_vector_type(16))) __bf16 v16bf;
typedef __attribute__((ext_vector_type(8)))  __bf16 v8bf;
typedef __attribute__((ext_vector_type(8)))  float  v8f;
typedef __attribute__((ext_vector_type(4)))  float  v4f;

// ---------------- prep: W_h fp32 -> bf16 (row-major, stays in 192MB L2) ------
__global__ __launch_bounds__(256)
void prep_wh(const float* __restrict__ wh, __bf16* __restrict__ wb) {
    int i = blockIdx.x * 256 + threadIdx.x;       // [0, H*H)
    wb[i] = (__bf16)wh[i];
}

// ---------------- prep: s[b][o] = dec[b] . W_s[o]  (32x512 outputs) ----------
__global__ __launch_bounds__(256)
void prep_s(const float* __restrict__ dec, const float* __restrict__ wsm,
            float* __restrict__ s) {
    int i = blockIdx.x * 256 + threadIdx.x;       // b*H + o
    int b = i >> 9;
    int o = i & (H_ - 1);
    const v4f* d = (const v4f*)(dec + (size_t)b * H_);
    const v4f* w = (const v4f*)(wsm + (size_t)o * H_);
    float acc = 0.f;
    for (int h = 0; h < H_ / 4; ++h) {
        v4f a = d[h], c = w[h];
        acc += a.x * c.x + a.y * c.y + a.z * c.z + a.w * c.w;
    }
    s[i] = acc;
}

// ---------------- main: e[b,t] = tanh(enc@W_h^T + s).v  (fused, WMMA) --------
__global__ __launch_bounds__(128)
void scores_kernel(const float* __restrict__ enc, const int* __restrict__ mask,
                   const __bf16* __restrict__ wb, const float* __restrict__ s,
                   const float* __restrict__ vvec, float* __restrict__ e) {
    extern __shared__ char smem_raw[];
    float* ldsf = (float*)smem_raw;               // [TTILE][LSTRF] f32

    const int b   = blockIdx.y;
    const int t0  = blockIdx.x * TTILE;
    const int tid = threadIdx.x;

    // ---- Stage enc tile [TTILE x H] f32 -> LDS via async DMA (no VGPR data path).
    // Dynamic LDS starts at offset 0 of this kernel's LDS window (no static __shared__).
    {
        const float* gsrc = enc + ((size_t)b * T_ + t0) * H_;
        for (int i = tid; i < TTILE * (H_ / 4); i += 128) {
            int row = i >> 7;                      // H_/4 == 128
            int c4  = i & 127;
            unsigned long long ga =
                (unsigned long long)(uintptr_t)(gsrc + (size_t)row * H_ + c4 * 4);
            unsigned int la = (unsigned int)(row * LSTRF + c4 * 4) * 4u;
            asm volatile("global_load_async_to_lds_b128 %0, %1, off"
                         :: "v"(la), "v"(ga) : "memory");
        }
        asm volatile("s_wait_asynccnt 0x0" ::: "memory");
    }
    __syncthreads();

    const int wave = tid >> 5;
    const int lane = tid & 31;
    const int half = lane >> 4;                    // K half-group (A & B layouts)
    const int nloc = lane & 15;                    // N within tile / A row within wave tile
    const int mrow = wave * 16 + nloc;             // this lane's A row in LDS tile

    // ---- Build all 16 A fragments once (register-resident for whole N loop).
    // A k order used here (16x32, B matches): lanes 0-15 hold K = k0+0..15,
    // lanes 16-31 hold K = k0+16..31.
    const float* aBaseF = ldsf + (size_t)mrow * LSTRF + half * 16;
    v16bf afr[16];
#pragma unroll
    for (int kk = 0; kk < 16; ++kk) {
        const float* p1 = aBaseF + kk * 32;        // run 1: K = 32kk + half*16
        const float* p2 = p1 + 8;                  // run 2: K = 32kk + 8 + half*16
        v4f a0 = *(const v4f*)(p1);
        v4f a1 = *(const v4f*)(p1 + 4);
        v4f b0 = *(const v4f*)(p2);
        v4f b1 = *(const v4f*)(p2 + 4);
        v16bf f;
        f[0]  = (__bf16)a0.x; f[1]  = (__bf16)a0.y; f[2]  = (__bf16)a0.z; f[3]  = (__bf16)a0.w;
        f[4]  = (__bf16)a1.x; f[5]  = (__bf16)a1.y; f[6]  = (__bf16)a1.z; f[7]  = (__bf16)a1.w;
        f[8]  = (__bf16)b0.x; f[9]  = (__bf16)b0.y; f[10] = (__bf16)b0.z; f[11] = (__bf16)b0.w;
        f[12] = (__bf16)b1.x; f[13] = (__bf16)b1.y; f[14] = (__bf16)b1.z; f[15] = (__bf16)b1.w;
        afr[kk] = f;
    }

    float epart[8];
#pragma unroll
    for (int r = 0; r < 8; ++r) epart[r] = 0.f;

    for (int n0 = 0; n0 < H_; n0 += 16) {
        // B fragment: 16 contiguous bf16 of row (n0+nloc); lanes 0-15 K lo-half,
        // lanes 16-31 K hi-half of each 32-wide K step.
        const __bf16* wrow = wb + (size_t)(n0 + nloc) * H_ + half * 16;
        if (n0 + 16 < H_)
            __builtin_prefetch(wb + (size_t)(n0 + 16 + nloc) * H_ + half * 16, 0, 3);

        v8f acc0 = {};
        v8f acc1 = {};
#pragma unroll
        for (int kk = 0; kk < 16; kk += 2) {       // two chains -> 2x matrix-pipe ILP
            v16bf bf0 = *(const v16bf*)(wrow + kk * 32);
            v16bf bf1 = *(const v16bf*)(wrow + kk * 32 + 32);
            acc0 = __builtin_amdgcn_wmma_f32_16x16x32_bf16(
                false, afr[kk],     false, bf0, (short)0, acc0, false, false);
            acc1 = __builtin_amdgcn_wmma_f32_16x16x32_bf16(
                false, afr[kk + 1], false, bf1, (short)0, acc1, false, false);
        }
        // Fused epilogue: tanh(h + s[n]) * v[n], accumulated over this N tile.
        float sn = s[b * H_ + n0 + nloc];
        float vn = vvec[n0 + nloc];
#pragma unroll
        for (int r = 0; r < 8; ++r)
            epart[r] += tanhf(acc0[r] + acc1[r] + sn) * vn;
    }

    // Reduce over the 16 lanes of each N-group (lanes 0-15: M=r, 16-31: M=8+r).
#pragma unroll
    for (int r = 0; r < 8; ++r) {
        float x = epart[r];
        for (int off = 1; off < 16; off <<= 1)
            x += __shfl_xor(x, off, 32);
        epart[r] = x;
    }

    if (nloc == 0) {
#pragma unroll
        for (int r = 0; r < 8; ++r) {
            int t = t0 + wave * 16 + half * 8 + r;
            float val = epart[r];
            e[(size_t)b * T_ + t] = mask[b * T_ + t] ? val : NEGV;
        }
    }
}

// ---------------- softmax stats: per-batch max and sum(exp) ------------------
__global__ __launch_bounds__(256)
void stats_kernel(const float* __restrict__ e, float* __restrict__ st) {
    __shared__ float red[256];
    int b = blockIdx.x, tid = threadIdx.x;
    const float* eb = e + (size_t)b * T_;

    float m = -3.4e38f;
    for (int t = tid; t < T_; t += 256) m = fmaxf(m, eb[t]);
    red[tid] = m; __syncthreads();
    for (int s2 = 128; s2 > 0; s2 >>= 1) {
        if (tid < s2) red[tid] = fmaxf(red[tid], red[tid + s2]);
        __syncthreads();
    }
    m = red[0]; __syncthreads();

    float sum = 0.f;
    for (int t = tid; t < T_; t += 256) sum += __expf(eb[t] - m);
    red[tid] = sum; __syncthreads();
    for (int s2 = 128; s2 > 0; s2 >>= 1) {
        if (tid < s2) red[tid] += red[tid + s2];
        __syncthreads();
    }
    if (tid == 0) { st[2 * b] = m; st[2 * b + 1] = red[0]; }
}

__global__ __launch_bounds__(256)
void zero_out(float* __restrict__ out) {
    out[blockIdx.x * 256 + threadIdx.x] = 0.f;
}

// ---------------- ctx[b,h] = sum_t softmax(e)[t] * enc[b,t,h] ----------------
__global__ __launch_bounds__(512)
void ctx_kernel(const float* __restrict__ enc, const float* __restrict__ e,
                const float* __restrict__ st, float* __restrict__ out) {
    __shared__ float sh_a[512];
    const int b  = blockIdx.y;
    const int t0 = blockIdx.x * 512;
    const int h  = threadIdx.x;

    float m   = st[2 * b];
    float inv = 1.0f / st[2 * b + 1];
    sh_a[h] = __expf(e[(size_t)b * T_ + t0 + h] - m) * inv;  // masked rows -> exp(-1e9)=0
    __syncthreads();

    const float* encb = enc + ((size_t)b * T_ + t0) * H_ + h;
    float acc = 0.f;
    for (int t = 0; t < 512; ++t)
        acc += sh_a[t] * encb[(size_t)t * H_];               // coalesced 2KB/row

    unsafeAtomicAdd(&out[b * H_ + h], acc);                  // global_atomic_add_f32
}

// ---------------- launch --------------------------------------------------
extern "C" void kernel_launch(void* const* d_in, const int* in_sizes, int n_in,
                              void* d_out, int out_size, void* d_ws, size_t ws_size,
                              hipStream_t stream) {
    const float* enc  = (const float*)d_in[0];   // [B,T,H]
    const int*   mask = (const int*)  d_in[1];   // [B,T]
    const float* dec  = (const float*)d_in[2];   // [B,H]
    const float* wh   = (const float*)d_in[3];   // [H,H]
    const float* wsm  = (const float*)d_in[4];   // [H,H]
    const float* vv   = (const float*)d_in[5];   // [H]
    float* out = (float*)d_out;                  // [B,H]

    char* wsb = (char*)d_ws;
    __bf16* wb = (__bf16*)wsb;                                   // 512 KB
    float*  s  = (float*)(wsb + 512 * 1024);                     // 64 KB
    float*  e  = (float*)(wsb + 512 * 1024 + 64 * 1024);         // 512 KB
    float*  st = (float*)(wsb + 512 * 1024 + 64 * 1024 + 512 * 1024);

    prep_wh<<<(H_ * H_) / 256, 256, 0, stream>>>(wh, wb);
    prep_s<<<(B_ * H_) / 256, 256, 0, stream>>>(dec, wsm, s);

    dim3 g1(T_ / TTILE, B_);
    size_t shmem = (size_t)TTILE * LSTRF * sizeof(float);        // ~129 KB (< 320 KB WGP LDS)
    scores_kernel<<<g1, 128, shmem, stream>>>(enc, mask, wb, s, vv, e);

    stats_kernel<<<B_, 256, 0, stream>>>(e, st);
    zero_out<<<(B_ * H_) / 256, 256, 0, stream>>>(out);

    dim3 g2(T_ / 512, B_);
    ctx_kernel<<<g2, 512, 0, stream>>>(enc, e, st, out);
}
